// MultiHeadAttention_41540923687042
// MI455X (gfx1250) — matrix operand order repaired and checked by reading, not compiled, not run on hardware
//
#include <hip/hip_runtime.h>

// ---------------------------------------------------------------------------
// MHA for B=4, S=1024, D=512, H=8 on MI455X (gfx1250, wave32).
// Pipeline: f32->bf16 cvt, WMMA GEMM projections, WMMA Q.K^T (scaled, f32),
// masked softmax (bf16 probs), WMMA attn.V, WMMA output proj + bias + residual.
// GEMM: 128x128 macro tile, BK=32, LDS double-buffered, software pipelined.
// Tile staging via Tensor Data Mover (tensor_load_to_lds, TENSORcnt) for all
// contiguous-K tiles; register+scatter transpose only for k-major B tiles.
// ---------------------------------------------------------------------------

typedef __attribute__((ext_vector_type(16))) __bf16        v16bf;
typedef __attribute__((ext_vector_type(8)))  __bf16        v8bf;
typedef __attribute__((ext_vector_type(4)))  __bf16        v4bf;
typedef __attribute__((ext_vector_type(8)))  float         v8f;
typedef __attribute__((ext_vector_type(4)))  float         f32x4;
typedef __attribute__((ext_vector_type(4)))  unsigned int  u32x4;
typedef __attribute__((ext_vector_type(8)))  int           i32x8;
typedef __attribute__((ext_vector_type(4)))  int           i32x4;

#define B_   4
#define S_   1024
#define D_   512
#define H_   8
#define HD_  4096      // H_*D_
#define BS_  4096      // B_*S_

enum { OUT_BF16 = 0, OUT_F32 = 1, OUT_FINAL = 2 };

// ---------------------------------------------------------------------------
// f32 -> bf16 conversion, 4 elems/thread (all counts are multiples of 4)
// ---------------------------------------------------------------------------
__global__ __launch_bounds__(256) void cvt_f32_bf16(const float* __restrict__ in,
                                                    __bf16* __restrict__ out, int n4) {
  int i = blockIdx.x * 256 + threadIdx.x;
  if (i < n4) {
    f32x4 f = ((const f32x4*)in)[i];
    v4bf o;
    o[0] = (__bf16)f[0]; o[1] = (__bf16)f[1]; o[2] = (__bf16)f[2]; o[3] = (__bf16)f[3];
    ((v4bf*)out)[i] = o;
  }
}

// ---------------------------------------------------------------------------
// TDM: DMA a 2D tile (32 elems contiguous x 128 rows, bf16, row stride
// `stride_elems`) from global memory into contiguous LDS at lds_off.
// D# per CDNA5 ISA ch.8: group0 {count=1, lds_addr, global_addr, type=2},
// group1 {data_size=2B, tensor dims maxed (tiles never OOB), tile 32x128,
// dim0 stride}, groups 2/3 zero (tile_dim3/4 = 0 -> unused).
// ---------------------------------------------------------------------------
__device__ __forceinline__ void tdm_load_tile(unsigned lds_off, const void* gptr,
                                              int stride_elems) {
  unsigned long long ga = (unsigned long long)gptr;
  u32x4 g0 = { 1u,                                        // count=1 (valid)
               lds_off,                                    // lds_addr
               (unsigned)ga,                               // global_addr[31:0]
               (unsigned)((ga >> 32) & 0x01FFFFFFu) | 0x80000000u }; // [56:32]|type=2
  i32x8 g1 = { (int)0x00010000,                            // data_size=1 (2 bytes)
               (int)0xFFFF0000u,                           // tensor_dim0 lo16 (max)
               (int)0xFFFF7FFFu,                           // td0 hi16 | td1 lo16
               (int)((32u << 16) | 0x7FFFu),               // td1 hi16 | tile_dim0=32
               128,                                        // tile_dim1=128, tile_dim2=0
               stride_elems,                               // tensor_dim0_stride lo32
               0, 0 };                                     // stride hi / dim1_stride
  i32x4 z4 = { 0, 0, 0, 0 };
  i32x8 z8 = { 0, 0, 0, 0, 0, 0, 0, 0 };
  __builtin_amdgcn_tensor_load_to_lds(g0, g1, z4, z4, z8, 0);
}

#define LDSOFF(p) ((unsigned)(unsigned long long)(const void*)(p))

// ---------------------------------------------------------------------------
// Tiled bf16 WMMA GEMM:  C[z] = scale * A[z] * B[z]  (+ bias + resid for FINAL)
//   A: M x K row-major (lda).  B: TRANSB ? B(k,n)=Bm[n*ldb+k] : B(k,n)=Bm[k*ldb+n]
//   256 threads = 8 waves (4 along M x 2 along N); each wave 2x4 WMMA tiles.
//   A tile (and B tile when TRANSB) staged by TDM; double buffered.
// ---------------------------------------------------------------------------
template <bool TRANSB, int MODE>
__global__ __launch_bounds__(256) void gemm_wmma(
    const __bf16* __restrict__ A,  int lda, long long aStrB, long long aStrH,
    const __bf16* __restrict__ Bm, int ldb, long long bStrB, long long bStrH,
    __bf16* __restrict__ Cb, float* __restrict__ Cf, int ldc,
    long long cStrB, long long cStrH,
    int K, int Hb, float scale,
    const float* __restrict__ bias, const float* __restrict__ resid) {
  __shared__ __attribute__((aligned(16))) __bf16 As[2][128][32];   // [buf][m][k]
  __shared__ __attribute__((aligned(16))) __bf16 Bs[2][128][32];   // [buf][n][k]

  const int z  = blockIdx.z;
  const int bb = z / Hb, hh = z % Hb;
  A  += (size_t)bb * aStrB + (size_t)hh * aStrH;
  Bm += (size_t)bb * bStrB + (size_t)hh * bStrH;
  const size_t cOff = (size_t)bb * cStrB + (size_t)hh * cStrH;

  const int tid   = threadIdx.x;
  const int lane  = tid & 31;
  const int wave  = tid >> 5;
  const int waveM = wave >> 1;      // 0..3, 32 rows each
  const int waveN = wave & 1;       // 0..1, 64 cols each
  const int half  = lane >> 4;      // lane group (ISA 7.12.2 layouts)
  const int l16   = lane & 15;
  const int m0 = blockIdx.x * 128;
  const int n0 = blockIdx.y * 128;

  // ---- workgroup-uniform tile pointers for TDM ----
  const __bf16* tA = A + (size_t)m0 * lda;                     // += 32 per K tile
  const __bf16* tB = TRANSB ? (Bm + (size_t)n0 * ldb) : nullptr;

  // ---- register path for k-major B tiles (transpose-scatter into [n][k]) ----
  const int kBn = tid >> 4;                  // k rows tid>>4 and +16
  const int nBn = (tid & 15) << 3;
  const __bf16* gB = Bm + (size_t)kBn * ldb + n0 + nBn;
  u32x4 b0, b1;
  auto fetchB = [&]() {
    b0 = *(const u32x4*)gB;
    b1 = *(const u32x4*)(gB + (size_t)16 * ldb);
    gB += (size_t)32 * ldb;
  };
  auto stageB = [&](int buf) {
    union { u32x4 q; v8bf h; } t0, t1;
    t0.q = b0; t1.q = b1;
#pragma unroll
    for (int i = 0; i < 8; ++i) {
      Bs[buf][nBn + i][kBn]      = t0.h[i];
      Bs[buf][nBn + i][kBn + 16] = t1.h[i];
    }
  };

  v8f acc[2][4];
#pragma unroll
  for (int mi = 0; mi < 2; ++mi)
#pragma unroll
    for (int ni = 0; ni < 4; ++ni) acc[mi][ni] = {};

  const int  ktc    = K >> 5;
  const bool issuer = (wave == 0);

  // ---- prologue: stage tile 0 ----
  if (issuer) {
    tdm_load_tile(LDSOFF(&As[0][0][0]), tA, lda);
    if (TRANSB) tdm_load_tile(LDSOFF(&Bs[0][0][0]), tB, ldb);
  }
  tA += 32;
  if (TRANSB) tB += 32;
  if (!TRANSB) { fetchB(); stageB(0); }

  int cur = 0;
  for (int kt = 0; kt < ktc; ++kt) {
    if (issuer) __builtin_amdgcn_s_wait_tensorcnt(0);  // tile kt landed in LDS
    __syncthreads();                                   // visible to all waves
    const bool more = (kt + 1 < ktc);
    if (more) {
      if (issuer) {                                    // DMA tile kt+1 (overlaps WMMA)
        tdm_load_tile(LDSOFF(&As[cur ^ 1][0][0]), tA, lda);
        if (TRANSB) tdm_load_tile(LDSOFF(&Bs[cur ^ 1][0][0]), tB, ldb);
      }
      tA += 32;
      if (TRANSB) tB += 32;
      if (!TRANSB) fetchB();                           // global loads overlap WMMA
    }

    // ---- fragments per ISA 7.12.2 (16-bit A 16x32 / B 32x16, wave32) ----
    union Frag { u32x4 q[2]; v16bf v; };
    Frag fa[2], fb[4];
#pragma unroll
    for (int mi = 0; mi < 2; ++mi) {
      const int row = waveM * 32 + mi * 16 + l16;
      fa[mi].q[0] = *(const u32x4*)&As[cur][row][half * 8];        // K 0..15 half
      fa[mi].q[1] = *(const u32x4*)&As[cur][row][16 + half * 8];   // K 16..31 half
    }
#pragma unroll
    for (int ni = 0; ni < 4; ++ni) {
      const int col = waveN * 64 + ni * 16 + l16;
      fb[ni].q[0] = *(const u32x4*)&Bs[cur][col][half * 8];        // K 0..15 half
      fb[ni].q[1] = *(const u32x4*)&Bs[cur][col][16 + half * 8];   // K 16..31 half
    }
#pragma unroll
    for (int mi = 0; mi < 2; ++mi)
#pragma unroll
      for (int ni = 0; ni < 4; ++ni)
        acc[mi][ni] = __builtin_amdgcn_wmma_f32_16x16x32_bf16(
            false, fa[mi].v, false, fb[ni].v, (short)0, acc[mi][ni], false, false);

    if (more) {
      if (!TRANSB) stageB(cur ^ 1);                    // scatter into other buffer
      cur ^= 1;
    }
  }

  // ---- epilogue: C/D layout -> row = half*8 + r, col = l16 ----
#pragma unroll
  for (int mi = 0; mi < 2; ++mi) {
#pragma unroll
    for (int ni = 0; ni < 4; ++ni) {
      const int col = n0 + waveN * 64 + ni * 16 + l16;
#pragma unroll
      for (int r = 0; r < 8; ++r) {
        const int row = m0 + waveM * 32 + mi * 16 + half * 8 + r;
        float vv = acc[mi][ni][r] * scale;
        const size_t idx = cOff + (size_t)row * ldc + col;
        if (MODE == OUT_BF16) {
          Cb[idx] = (__bf16)vv;
        } else if (MODE == OUT_F32) {
          Cf[idx] = vv;
        } else {
          Cf[idx] = vv + bias[col] + resid[(size_t)row * ldc + col];
        }
      }
    }
  }
}

// ---------------------------------------------------------------------------
// Masked softmax over one row of scores (S=1024). One block (256 thr) per row.
// scores: [B,H,S,S] f32 (already scaled); mask: [B,S,S] i32; attn: bf16 out.
// ---------------------------------------------------------------------------
__global__ __launch_bounds__(256) void softmax_mask_kernel(
    const float* __restrict__ scores, const int* __restrict__ mask,
    __bf16* __restrict__ attn) {
  __shared__ float red[256];
  const int r  = blockIdx.x;            // 0 .. B*H*S-1
  const int q  = r & (S_ - 1);
  const int bh = r >> 10;
  const int b  = bh >> 3;               // / H_
  const float* srow = scores + (size_t)r * S_;
  const int*   mrow = mask + ((size_t)b * S_ + q) * S_;
  __bf16*      arow = attn + (size_t)r * S_;
  const int t = threadIdx.x;

  float v[4];
  float mx = -3.0e38f;
#pragma unroll
  for (int i = 0; i < 4; ++i) {
    int k = t + i * 256;
    float s = (mrow[k] == 0) ? -1.0e6f : srow[k];
    v[i] = s;
    mx = fmaxf(mx, s);
  }
  red[t] = mx; __syncthreads();
  for (int s2 = 128; s2 > 0; s2 >>= 1) {
    if (t < s2) red[t] = fmaxf(red[t], red[t + s2]);
    __syncthreads();
  }
  mx = red[0]; __syncthreads();

  float sum = 0.f;
#pragma unroll
  for (int i = 0; i < 4; ++i) { v[i] = __expf(v[i] - mx); sum += v[i]; }
  red[t] = sum; __syncthreads();
  for (int s2 = 128; s2 > 0; s2 >>= 1) {
    if (t < s2) red[t] += red[t + s2];
    __syncthreads();
  }
  const float inv = 1.f / red[0];
#pragma unroll
  for (int i = 0; i < 4; ++i) arow[t + i * 256] = (__bf16)(v[i] * inv);
}

// ---------------------------------------------------------------------------
extern "C" void kernel_launch(void* const* d_in, const int* in_sizes, int n_in,
                              void* d_out, int out_size, void* d_ws, size_t ws_size,
                              hipStream_t stream) {
  (void)in_sizes; (void)n_in; (void)out_size; (void)ws_size;
  const float* query = (const float*)d_in[0];
  const float* key   = (const float*)d_in[1];
  const float* value = (const float*)d_in[2];
  const int*   mask  = (const int*)d_in[3];
  const float* Wq    = (const float*)d_in[4];
  const float* Wk    = (const float*)d_in[5];
  const float* Wv    = (const float*)d_in[6];
  const float* Wo    = (const float*)d_in[7];
  const float* bo    = (const float*)d_in[8];
  float* out = (float*)d_out;

  size_t off = 0;
  auto take = [&](size_t nbytes) -> void* {
    void* p = (void*)((char*)d_ws + off);
    off += (nbytes + 255) & ~(size_t)255;
    return p;
  };
  __bf16* qb  = (__bf16*)take((size_t)BS_ * D_ * 2);
  __bf16* kb  = (__bf16*)take((size_t)BS_ * D_ * 2);
  __bf16* vb  = (__bf16*)take((size_t)BS_ * D_ * 2);
  __bf16* Wqb = (__bf16*)take((size_t)D_ * HD_ * 2);
  __bf16* Wkb = (__bf16*)take((size_t)D_ * HD_ * 2);
  __bf16* Wvb = (__bf16*)take((size_t)D_ * HD_ * 2);
  __bf16* Wob = (__bf16*)take((size_t)HD_ * D_ * 2);
  __bf16* Qp  = (__bf16*)take((size_t)BS_ * HD_ * 2);   // [B,S,H*D]
  __bf16* Kp  = (__bf16*)take((size_t)BS_ * HD_ * 2);
  __bf16* Vp  = (__bf16*)take((size_t)BS_ * HD_ * 2);
  float*  sc  = (float*)take((size_t)B_ * H_ * S_ * S_ * 4);  // scores f32
  __bf16* at  = (__bf16*)take((size_t)B_ * H_ * S_ * S_ * 2); // attn bf16
  __bf16* cx  = (__bf16*)take((size_t)BS_ * HD_ * 2);         // ctx concat [B,S,H*D]

  const dim3 blk(256);
  auto cvt = [&](const float* src, __bf16* dst, size_t n) {
    int n4 = (int)(n / 4);
    cvt_f32_bf16<<<dim3((n4 + 255) / 256), blk, 0, stream>>>(src, dst, n4);
  };
  cvt(query, qb, (size_t)BS_ * D_);
  cvt(key,   kb, (size_t)BS_ * D_);
  cvt(value, vb, (size_t)BS_ * D_);
  cvt(Wq, Wqb, (size_t)D_ * HD_);
  cvt(Wk, Wkb, (size_t)D_ * HD_);
  cvt(Wv, Wvb, (size_t)D_ * HD_);
  cvt(Wo, Wob, (size_t)HD_ * D_);

  // Projections: [BS_,D_] x [D_,HD_] -> bf16 [B,S,H*D]
  {
    dim3 g(BS_ / 128, HD_ / 128, 1);
    gemm_wmma<false, OUT_BF16><<<g, blk, 0, stream>>>(
        qb, D_, 0, 0, Wqb, HD_, 0, 0, Qp, nullptr, HD_, 0, 0, D_, 1, 1.f, nullptr, nullptr);
    gemm_wmma<false, OUT_BF16><<<g, blk, 0, stream>>>(
        kb, D_, 0, 0, Wkb, HD_, 0, 0, Kp, nullptr, HD_, 0, 0, D_, 1, 1.f, nullptr, nullptr);
    gemm_wmma<false, OUT_BF16><<<g, blk, 0, stream>>>(
        vb, D_, 0, 0, Wvb, HD_, 0, 0, Vp, nullptr, HD_, 0, 0, D_, 1, 1.f, nullptr, nullptr);
  }
  // Scores: per (b,h)  Q[S,D] * K^T -> f32 [B,H,S,S], scaled by 1/sqrt(D)
  {
    dim3 g(S_ / 128, S_ / 128, B_ * H_);
    gemm_wmma<true, OUT_F32><<<g, blk, 0, stream>>>(
        Qp, HD_, (long long)S_ * HD_, D_,
        Kp, HD_, (long long)S_ * HD_, D_,
        nullptr, sc, S_, (long long)H_ * S_ * S_, (long long)S_ * S_,
        D_, H_, 0.04419417382415922f, nullptr, nullptr);
  }
  // Masked softmax -> bf16 attn
  softmax_mask_kernel<<<dim3(B_ * H_ * S_), blk, 0, stream>>>(sc, mask, at);
  // Context: per (b,h)  attn[S,S] * V[S,D] -> bf16 ctx stored as [B,S,H*D]
  {
    dim3 g(S_ / 128, D_ / 128, B_ * H_);
    gemm_wmma<false, OUT_BF16><<<g, blk, 0, stream>>>(
        at, S_, (long long)H_ * S_ * S_, (long long)S_ * S_,
        Vp, HD_, (long long)S_ * HD_, D_,
        cx, nullptr, HD_, (long long)S_ * HD_, D_,
        S_, H_, 1.f, nullptr, nullptr);
  }
  // Output projection + bias + residual -> f32 d_out
  {
    dim3 g(BS_ / 128, D_ / 128, 1);
    gemm_wmma<false, OUT_FINAL><<<g, blk, 0, stream>>>(
        cx, HD_, 0, 0, Wob, D_, 0, 0,
        nullptr, out, D_, 0, 0, HD_, 1, 1.f, bo, query);
  }
}
